// BrainLayer_13486197309468
// MI455X (gfx1250) — matrix operand order repaired and checked by reading, not compiled, not run on hardware
//
#include <hip/hip_runtime.h>
#include <math.h>

// ---------------------------------------------------------------------------
// Echo-state reservoir on gfx1250 (MI455X): persistent-kernel bf16 WMMA path
// with async global->LDS staging (ASYNCcnt) overlapped with input-proj WMMAs.
//   B=16 (== WMMA M), T=1024, F=128, R=2048, HALF=1024, GAMMA=0.95
//   h_new = (1-g)*h + g*tanh(h @ Wrec^T + [x_t @ Win^T | 0] + bias)
// ---------------------------------------------------------------------------

typedef __attribute__((ext_vector_type(16))) __bf16 v16bf;
typedef __attribute__((ext_vector_type(8)))  __bf16 v8bf;
typedef __attribute__((ext_vector_type(8)))  float  v8f;

#define B_     16
#define T_     1024
#define F_     128
#define R_     2048
#define HALF_  1024
#define GAMMA_ 0.95f
#define NWG_   16

// Async DMA: memory -> LDS, 128 bits, no VGPR data path (tracked by ASYNCcnt).
__device__ __forceinline__ void async_ld_b128(unsigned lds_off,
                                              const void* gaddr) {
  asm volatile("global_load_async_to_lds_b128 %0, %1, off"
               :: "v"(lds_off),
                  "v"((unsigned long long)(uintptr_t)gaddr)
               : "memory");
}
__device__ __forceinline__ void wait_asynccnt0() {
  asm volatile("s_wait_asynccnt 0x0" ::: "memory");
}

// ---------------------------------------------------------------------------
// Pre-pack Wrec [R,R] (out[b,n] = sum_k h[b,k]*W[n,k]) into the CDNA5 16-bit
// B-matrix (32x16) per-lane fragment layout:
//   lanes 0-15 : N = n0+lane,    K = k0 + j (j<8), k0 + 16 + (j-8) (j>=8)
//   lanes 16-31: N = n0+lane-16, K = k0 + 8 + j (j<8), k0 + 24 + (j-8) (j>=8)
// Flat index p = (((tn*64)+tk)*32 + lane)*16 + j -> each lane's 16 bf16 are
// contiguous (one v16bf / 2x b128 load in the hot loop).
// ---------------------------------------------------------------------------
__global__ void pack_wrec_kernel(const float* __restrict__ W,
                                 __bf16* __restrict__ Wp) {
  int p  = blockIdx.x * 256 + threadIdx.x;      // 0 .. R_*R_-1
  int j  = p & 15;
  int l  = (p >> 4) & 31;
  int tk = (p >> 9) & 63;
  int tn = p >> 15;
  int n  = tn * 16 + (l & 15);
  int k  = tk * 32 + ((j < 8) ? 0 : 8) + ((l < 16) ? 0 : 8) + j;
  Wp[p] = (__bf16)W[(size_t)n * R_ + k];
}

// Same packing for Win rows 0..HALF-1 (the in_cor mask kills the rest), K=F.
__global__ void pack_win_kernel(const float* __restrict__ W,
                                __bf16* __restrict__ Wp) {
  int p  = blockIdx.x * 256 + threadIdx.x;      // 0 .. HALF_*F_-1
  int j  = p & 15;
  int l  = (p >> 4) & 31;
  int tk = (p >> 9) & 3;                        // F_/32 = 4
  int tn = p >> 11;                             // HALF_/16 = 64 tiles
  int n  = tn * 16 + (l & 15);
  int k  = tk * 32 + ((j < 8) ? 0 : 8) + ((l < 16) ? 0 : 8) + j;
  Wp[p] = (__bf16)W[(size_t)n * F_ + k];
}

__global__ void cvt_inputs_kernel(const float* __restrict__ x,
                                  __bf16* __restrict__ xb) {
  int i = blockIdx.x * 256 + threadIdx.x;       // B_*T_*F_ elements
  xb[i] = (__bf16)x[i];
}

__global__ void init_h_kernel(const float* __restrict__ rs,
                              float* __restrict__ hf,
                              __bf16* __restrict__ hb,
                              unsigned* __restrict__ sync_cnt) {
  int i = blockIdx.x * 256 + threadIdx.x;       // B_*R_ elements
  if (i == 0) *sync_cnt = 0u;                   // reset grid barrier each call
  int r = i & (R_ - 1);
  float v = rs[r];
  hf[i] = v;
  hb[i] = (__bf16)v;
}

// ---------------------------------------------------------------------------
// Persistent step kernel. Grid: 16 WGs x 256 threads (8 wave32). Each wave
// owns one 16x16 output tile (16 batch rows x 16 reservoir columns) for all
// 1024 steps. Full bf16 h (64 KB) is async-DMA'd into LDS each step, hidden
// behind the global-only input-projection WMMAs. The K loop is fully
// unrolled so every fragment load is base-pointer + immediate offset.
// ---------------------------------------------------------------------------
__global__ __launch_bounds__(256) void reservoir_kernel(
    float*        __restrict__ hf0,
    float*        __restrict__ hf1,
    __bf16*       __restrict__ hb0,
    __bf16*       __restrict__ hb1,
    const __bf16* __restrict__ Wp,     // packed Wrec [128 ntiles][64 ktiles][32][16]
    const __bf16* __restrict__ WinP,   // packed Win  [64  ntiles][4  ktiles][32][16]
    const __bf16* __restrict__ Xb,     // bf16 inputs [B][T][F]
    const float*  __restrict__ bias,   // [R]
    float*        __restrict__ out,    // [B][T][HALF]
    unsigned*     __restrict__ sync_cnt) {
  __shared__ __bf16 lds_h[B_ * R_];    // 16*2048*2 = 65536 bytes

  // Raw 32-bit LDS byte offset (generic LDS address truncates to the offset).
  const unsigned lds_base = (unsigned)(uintptr_t)&lds_h[0];

  const int lane   = threadIdx.x & 31;
  const int wave   = threadIdx.x >> 5;
  const int n_tile = blockIdx.x * 8 + wave;     // 0..127
  const int n0     = n_tile * 16;

  // 16-bit A-matrix (16x32) per-lane layout: lane m (m = lane&15) holds
  // K runs [k0+koff, +8) and [k0+koff+16, +8), koff = 0 (lanes<16) or 8.
  const int arow = lane & 15;
  const int koff = (lane < 16) ? 0 : 8;

  // C/D 16x16 f32 layout: VGPR r -> M = r + 8*(lane>=16), N = n0 + (lane&15).
  const int   n    = n0 + (lane & 15);
  const int   mb   = (lane < 16) ? 0 : 8;
  const float bn   = bias[n];
  const bool  wout = (n >= HALF_);

  // Hoisted per-wave/per-lane fragment base pointers (loop-invariant).
  const __bf16* wp_base  = Wp   + ((size_t)n_tile * 64 * 32 + lane) * 16;
  const __bf16* win_base = WinP + ((size_t)n_tile * 4 * 32 + lane) * 16;
  const __bf16* x_base   = Xb   + (size_t)arow * T_ * F_ + koff;
  const __bf16* ldsa     = &lds_h[arow * R_ + koff];

  float*  hfs[2] = {hf0, hf1};
  __bf16* hbs[2] = {hb0, hb1};

  for (int t = 0; t < T_; ++t) {
    const int     c      = t & 1;
    const float*  hcur_f = hfs[c];
    const __bf16* hcur_b = hbs[c];
    float*        hnxt_f = hfs[c ^ 1];
    __bf16*       hnxt_b = hbs[c ^ 1];

    // ---- kick off async DMA of h into LDS (16 x b128 per thread) ----------
    {
      const char* src = (const char*)hcur_b;
      #pragma unroll
      for (int i = 0; i < 16; ++i) {
        const int idx = (int)threadIdx.x + i * 256;   // 4096 x 16B = 64 KB
        async_ld_b128(lds_base + (unsigned)idx * 16u, src + (size_t)idx * 16);
      }
    }

    v8f acc = {};

    // ---- input projection (global-only reads) overlaps the staging DMA ----
    if (n0 < HALF_) {
      const __bf16* xrow = x_base + (size_t)t * F_;
      #pragma unroll
      for (int kt = 0; kt < F_ / 32; ++kt) {
        v8bf alo = *(const v8bf*)(xrow + kt * 32);
        v8bf ahi = *(const v8bf*)(xrow + kt * 32 + 16);
        v16bf a = __builtin_shufflevector(alo, ahi,
            0, 1, 2, 3, 4, 5, 6, 7, 8, 9, 10, 11, 12, 13, 14, 15);
        v16bf b = *(const v16bf*)(win_base + kt * 512);
        acc = __builtin_amdgcn_wmma_f32_16x16x32_bf16(
            false, a, false, b, (short)0, acc, false, false);
      }
    }

    wait_asynccnt0();                 // this thread's DMA into LDS is done
    __syncthreads();                  // whole WG's staging visible

    // ---- recurrent GEMM: 64 x v_wmma_f32_16x16x32_bf16, f32 accumulate ----
    // Fully unrolled: ds_load_b128 / global_load_b128 with immediate offsets
    // off loop-invariant base pointers; no per-iteration address VALU.
    #pragma unroll
    for (int kt = 0; kt < R_ / 32; ++kt) {
      v8bf alo = *(const v8bf*)(ldsa + kt * 32);        // offset kt*64  B
      v8bf ahi = *(const v8bf*)(ldsa + kt * 32 + 16);   // offset +32    B
      v16bf a = __builtin_shufflevector(alo, ahi,
          0, 1, 2, 3, 4, 5, 6, 7, 8, 9, 10, 11, 12, 13, 14, 15);
      v16bf b = *(const v16bf*)(wp_base + kt * 512);    // offset kt*1024 B
      acc = __builtin_amdgcn_wmma_f32_16x16x32_bf16(
          false, a, false, b, (short)0, acc, false, false);
    }

    // ---- epilogue: leaky-tanh update, dual-precision h store, output tap ---
    #pragma unroll
    for (int r = 0; r < 8; ++r) {
      int   m    = mb + r;
      float hold = hcur_f[m * R_ + n];
      float hn   = (1.0f - GAMMA_) * hold + GAMMA_ * tanhf(acc[r] + bn);
      hnxt_f[m * R_ + n] = hn;
      hnxt_b[m * R_ + n] = (__bf16)hn;
      if (wout) out[((size_t)m * T_ + t) * HALF_ + (n - HALF_)] = hn;
    }

    // ---- device-wide release/acquire barrier between steps ----------------
    __threadfence();                  // order this thread's h stores (release)
    __syncthreads();                  // whole WG done + fenced; LDS reusable
    if (threadIdx.x == 0) {
      __hip_atomic_fetch_add(sync_cnt, 1u, __ATOMIC_ACQ_REL,
                             __HIP_MEMORY_SCOPE_AGENT);
      const unsigned target = (unsigned)(t + 1) * NWG_;
      while (__hip_atomic_load(sync_cnt, __ATOMIC_ACQUIRE,
                               __HIP_MEMORY_SCOPE_AGENT) < target)
        __builtin_amdgcn_s_sleep(1);
    }
    __syncthreads();                  // all WGs arrived; h[next] visible
  }
}

// ---------------------------------------------------------------------------
extern "C" void kernel_launch(void* const* d_in, const int* in_sizes, int n_in,
                              void* d_out, int out_size, void* d_ws, size_t ws_size,
                              hipStream_t stream) {
  const float* x    = (const float*)d_in[0];   // [B,T,F]
  const float* Win  = (const float*)d_in[1];   // [R,F]
  const float* Wrec = (const float*)d_in[2];   // [R,R]
  const float* bias = (const float*)d_in[3];   // [R]
  const float* rs   = (const float*)d_in[4];   // [R]
  float* out = (float*)d_out;                  // [B,T,HALF]

  // Workspace carve-up (~12.8 MB total).
  char* ws = (char*)d_ws;
  float*    hf0  = (float*)ws;    ws += (size_t)B_ * R_ * sizeof(float);
  float*    hf1  = (float*)ws;    ws += (size_t)B_ * R_ * sizeof(float);
  __bf16*   hb0  = (__bf16*)ws;   ws += (size_t)B_ * R_ * sizeof(__bf16);
  __bf16*   hb1  = (__bf16*)ws;   ws += (size_t)B_ * R_ * sizeof(__bf16);
  __bf16*   Wp   = (__bf16*)ws;   ws += (size_t)R_ * R_ * sizeof(__bf16);
  __bf16*   WinP = (__bf16*)ws;   ws += (size_t)HALF_ * F_ * sizeof(__bf16);
  __bf16*   Xb   = (__bf16*)ws;   ws += (size_t)B_ * T_ * F_ * sizeof(__bf16);
  unsigned* cnt  = (unsigned*)ws; ws += 256;   // grid barrier counter

  pack_wrec_kernel<<<(R_ * R_) / 256, 256, 0, stream>>>(Wrec, Wp);
  pack_win_kernel<<<(HALF_ * F_) / 256, 256, 0, stream>>>(Win, WinP);
  cvt_inputs_kernel<<<(B_ * T_ * F_) / 256, 256, 0, stream>>>(x, Xb);
  init_h_kernel<<<(B_ * R_) / 256, 256, 0, stream>>>(rs, hf0, hb0, cnt);

  reservoir_kernel<<<NWG_, 256, 0, stream>>>(hf0, hf1, hb0, hb1,
                                             Wp, WinP, Xb, bias, out, cnt);
}
